// S32DBS_9371618640304
// MI455X (gfx1250) — compile-verified, matrix operand order checked by reading, not run on hardware
//
#include <hip/hip_runtime.h>
#include <hip/hip_bf16.h>
#include <math.h>

#define BB  2
#define HH  64
#define WWD 64
#define DM  192
#define DI  384
#define DSN 16
#define DTR 12
#define LL  4096
#define NCH 32            // scan chunks
#define CHUNK (LL / NCH)  // 128

typedef __attribute__((ext_vector_type(16))) _Float16 v16h;
typedef __attribute__((ext_vector_type(8)))  _Float16 v8h;
typedef __attribute__((ext_vector_type(4)))  _Float16 v4h;
typedef __attribute__((ext_vector_type(8)))  float    v8f;

__device__ __forceinline__ float silu_f(float x) {
    return x * __builtin_amdgcn_rcpf(1.0f + __expf(-x));
}

// WMMA 16-bit fragment from a row-major f16 row: two contiguous b128 loads.
// lane<16 -> K in {k+0..7, k+16..23}; lane>=16 -> {k+8..15, k+24..31}.
__device__ __forceinline__ v16h frag_h(const _Float16* __restrict__ row, int k, int lane) {
    int koff = (lane & 16) ? 8 : 0;
    v8h lo = *(const v8h*)(row + k + koff);
    v8h hi = *(const v8h*)(row + k + koff + 16);
    v16h f;
    #pragma unroll
    for (int i = 0; i < 8; ++i) { f[i] = lo[i]; f[8 + i] = hi[i]; }
    return f;
}

template<int MODE>
__device__ __forceinline__ void store_tile(const v8f& acc, int m0, int n0, int lane,
                                           float* __restrict__ out0, float* __restrict__ out1)
{
    int col   = n0 + (lane & 15);
    int rbase = m0 + ((lane >> 4) << 3);
    #pragma unroll
    for (int j = 0; j < 8; ++j) {
        int m = rbase + j;
        float v = acc[j];
        if (MODE == 0) {
            int b = m >> 12, l = m & (LL - 1);
            if (col < DI) out0[(((size_t)(b * DI + col)) << 12) + l] = v;   // xa NCHW
            else          out1[(size_t)m * DI + (col - DI)] = silu_f(v);    // z
        } else if (MODE == 1) {
            int b = m >> 12, l = m & (LL - 1);
            out0[(((size_t)(b * DI + col)) << 12) + l] = v;                 // c NCHW
        } else if (MODE == 2) {
            out0[(size_t)m * 48 + col] = v;                                 // xdbl (48-padded)
        } else {
            out0[(size_t)m * DM + col] = v;                                 // final out
        }
    }
}

// C = A(MxK) @ Bw(NxK)^T, f16 operands, one wave per 64x16 tile
// (4 M-tiles register-blocked, B fragment reused 4x, double-buffered k-loop).
template<int MODE, int K, int LDA, int LDB>
__global__ void wmma_gemm_kernel(const _Float16* __restrict__ A, const _Float16* __restrict__ Bw,
                                 float* __restrict__ out0, float* __restrict__ out1,
                                 int Mblks, int Ntiles)
{
    int wave = blockIdx.x * (blockDim.x >> 5) + (threadIdx.x >> 5);
    if (wave >= Mblks * Ntiles) return;          // uniform per-wave, EXEC stays all-1
    int lane = threadIdx.x & 31;
    int mb = wave % Mblks;
    int nt = wave / Mblks;
    int m0 = mb << 6;                            // 64 rows per wave
    int n0 = nt << 4;

    const _Float16* Brow  = Bw + (size_t)(n0 + (lane & 15)) * LDB;
    const _Float16* Arow0 = A + (size_t)(m0 + (lane & 15)) * LDA;
    const _Float16* Arow1 = Arow0 + 16 * LDA;
    const _Float16* Arow2 = Arow0 + 32 * LDA;
    const _Float16* Arow3 = Arow0 + 48 * LDA;

    v8f acc0 = {0.f,0.f,0.f,0.f,0.f,0.f,0.f,0.f};
    v8f acc1 = acc0, acc2 = acc0, acc3 = acc0;

    // software pipeline: fragments for step k live while step k+32 loads issue
    v16h bf = frag_h(Brow,  0, lane);
    v16h a0 = frag_h(Arow0, 0, lane);
    v16h a1 = frag_h(Arow1, 0, lane);
    v16h a2 = frag_h(Arow2, 0, lane);
    v16h a3 = frag_h(Arow3, 0, lane);

    for (int k = 32; k < K; k += 32) {
        v16h nbf = frag_h(Brow,  k, lane);
        v16h na0 = frag_h(Arow0, k, lane);
        v16h na1 = frag_h(Arow1, k, lane);
        v16h na2 = frag_h(Arow2, k, lane);
        v16h na3 = frag_h(Arow3, k, lane);
        acc0 = __builtin_amdgcn_wmma_f32_16x16x32_f16(false, a0, false, bf, (short)0, acc0, false, false);
        acc1 = __builtin_amdgcn_wmma_f32_16x16x32_f16(false, a1, false, bf, (short)0, acc1, false, false);
        acc2 = __builtin_amdgcn_wmma_f32_16x16x32_f16(false, a2, false, bf, (short)0, acc2, false, false);
        acc3 = __builtin_amdgcn_wmma_f32_16x16x32_f16(false, a3, false, bf, (short)0, acc3, false, false);
        bf = nbf; a0 = na0; a1 = na1; a2 = na2; a3 = na3;
    }
    acc0 = __builtin_amdgcn_wmma_f32_16x16x32_f16(false, a0, false, bf, (short)0, acc0, false, false);
    acc1 = __builtin_amdgcn_wmma_f32_16x16x32_f16(false, a1, false, bf, (short)0, acc1, false, false);
    acc2 = __builtin_amdgcn_wmma_f32_16x16x32_f16(false, a2, false, bf, (short)0, acc2, false, false);
    acc3 = __builtin_amdgcn_wmma_f32_16x16x32_f16(false, a3, false, bf, (short)0, acc3, false, false);

    store_tile<MODE>(acc0, m0,      n0, lane, out0, out1);
    store_tile<MODE>(acc1, m0 + 16, n0, lane, out0, out1);
    store_tile<MODE>(acc2, m0 + 32, n0, lane, out0, out1);
    store_tile<MODE>(acc3, m0 + 48, n0, lane, out0, out1);
}

// f32 -> f16 pack, 4 elements/thread (b128 in, b64 out)
__global__ void pack_f16_v4(const float* __restrict__ in, _Float16* __restrict__ out, int n4)
{
    int i = blockIdx.x * blockDim.x + threadIdx.x;
    if (i >= n4) return;
    float4 v = ((const float4*)in)[i];
    v4h o;
    o[0] = (_Float16)v.x; o[1] = (_Float16)v.y; o[2] = (_Float16)v.z; o[3] = (_Float16)v.w;
    ((v4h*)out)[i] = o;
}

// x_proj_w (44x384) -> f16 zero-padded (48x384)
__global__ void pack_pad_xproj(const float* __restrict__ in, _Float16* __restrict__ out)
{
    int i = blockIdx.x * blockDim.x + threadIdx.x;
    if (i >= 48 * DI) return;
    int r = i / DI;
    out[i] = (r < 44) ? (_Float16)in[i] : (_Float16)0.0f;
}

// Fused depthwise 3x3 conv (SAME) + bias + SiLU for both branches, with the
// scan_path gather. Emits u=(b,d,l') f32 and s=(b*l', d) = xs+cs as f16.
__global__ void conv_scan_kernel(const float* __restrict__ xa_t, const float* __restrict__ con_t,
                                 const float* __restrict__ cw,  const float* __restrict__ cb,
                                 const float* __restrict__ ccw, const float* __restrict__ ccb,
                                 const int* __restrict__ scan_path,
                                 float* __restrict__ u, _Float16* __restrict__ s)
{
    int idx = blockIdx.x * blockDim.x + threadIdx.x;
    if (idx >= BB * DI * LL) return;
    int l  = idx & (LL - 1);
    int bd = idx >> 12;
    int d  = bd % DI;
    int b  = bd / DI;
    int p  = scan_path[l];
    int h  = p >> 6, w = p & (WWD - 1);
    const float* xin = xa_t + ((size_t)bd << 12);
    const float* cin = con_t + ((size_t)bd << 12);
    float ax = cb[d], ac = ccb[d];
    #pragma unroll
    for (int kh = 0; kh < 3; ++kh) {
        int hh = h + kh - 1;
        if ((unsigned)hh >= (unsigned)HH) continue;
        #pragma unroll
        for (int kw = 0; kw < 3; ++kw) {
            int ww2 = w + kw - 1;
            if ((unsigned)ww2 >= (unsigned)WWD) continue;
            float wx = cw [d * 9 + kh * 3 + kw];
            float wc = ccw[d * 9 + kh * 3 + kw];
            ax += wx * xin[(hh << 6) + ww2];
            ac += wc * cin[(hh << 6) + ww2];
        }
    }
    float sx = silu_f(ax), sc = silu_f(ac);
    u[idx] = sx;
    int m = (b << 12) + l;
    s[(size_t)m * DI + d] = (_Float16)(sx + sc);
}

// delta = softplus(dt_proj_w @ xdbl[:, :12] + dt_proj_b), stored (b,d,l)
__global__ void dt_kernel(const float* __restrict__ xdbl, const float* __restrict__ dtw,
                          const float* __restrict__ dtb, float* __restrict__ delta)
{
    int idx = blockIdx.x * blockDim.x + threadIdx.x;
    if (idx >= BB * DI * LL) return;
    int l  = idx & (LL - 1);
    int bd = idx >> 12;
    int d  = bd % DI;
    int b  = bd / DI;
    const float* row = xdbl + ((size_t)((b << 12) + l)) * 48;
    float acc = dtb[d];
    #pragma unroll
    for (int r = 0; r < DTR; ++r) acc += row[r] * dtw[d * DTR + r];
    delta[idx] = (acc > 20.0f) ? acc : log1pf(__expf(acc));
}

// --- Chunked selective scan: h_l = a_l*h_{l-1} + b_l per (chain, state n) ---
__global__ void scan_chunk_reduce(const float* __restrict__ u, const float* __restrict__ delta,
                                  const float* __restrict__ xdbl, const float* __restrict__ A_logs,
                                  float* __restrict__ Pbuf, float* __restrict__ Sbuf)
{
    int tid = threadIdx.x;
    int g = blockIdx.x * (blockDim.x >> 4) + (tid >> 4);
    if (g >= BB * DI * NCH) return;
    int n = tid & 15;
    int chain = g >> 5;                 // NCH == 32
    int c = g & (NCH - 1);
    int b = chain / DI, d = chain % DI;
    float Adn = -__expf(A_logs[d * DSN + n]);
    size_t base = (size_t)chain * LL + (size_t)c * CHUNK;
    size_t xbase = ((size_t)b * LL + (size_t)c * CHUNK) * 48;
    float P = 1.0f, S = 0.0f;
    for (int i = 0; i < CHUNK; ++i) {
        float dv = delta[base + i];
        float uv = u[base + i];
        float Bv = xdbl[xbase + (size_t)i * 48 + DTR + n];
        float dA = __expf(dv * Adn);
        S = dA * S + dv * Bv * uv;
        P *= dA;
    }
    int o = (g << 4) + n;
    Pbuf[o] = P;
    Sbuf[o] = S;
}

__global__ void scan_combine(const float* __restrict__ Pbuf, const float* __restrict__ Sbuf,
                             float* __restrict__ Hbuf)
{
    int idx = blockIdx.x * blockDim.x + threadIdx.x;
    if (idx >= BB * DI * DSN) return;
    int chain = idx >> 4;
    int n = idx & 15;
    float h = 0.0f;
    for (int c = 0; c < NCH; ++c) {
        int o = (((chain << 5) + c) << 4) + n;
        Hbuf[o] = h;
        h = Pbuf[o] * h + Sbuf[o];
    }
}

// Pass 3: re-scan within chunk from recovered state; scatter y directly to the
// spatial (rev-gathered) position in (b, l, d) layout -> LN reads coalesced.
__global__ void scan_apply(const float* __restrict__ u, const float* __restrict__ delta,
                           const float* __restrict__ xdbl, const float* __restrict__ A_logs,
                           const float* __restrict__ Ds, const float* __restrict__ Hbuf,
                           const int* __restrict__ scan_path, float* __restrict__ yt)
{
    int tid = threadIdx.x;
    int g = blockIdx.x * (blockDim.x >> 4) + (tid >> 4);
    if (g >= BB * DI * NCH) return;
    int n = tid & 15;
    int chain = g >> 5;
    int c = g & (NCH - 1);
    int b = chain / DI, d = chain % DI;
    float Adn = -__expf(A_logs[d * DSN + n]);
    float Dd  = Ds[d];
    size_t base = (size_t)chain * LL + (size_t)c * CHUNK;
    int    lbase = c * CHUNK;
    size_t xbase = ((size_t)b * LL + (size_t)c * CHUNK) * 48;
    float h = Hbuf[(g << 4) + n];
    for (int i = 0; i < CHUNK; ++i) {
        float dv = delta[base + i];
        float uv = u[base + i];
        float Bv = xdbl[xbase + (size_t)i * 48 + DTR + n];
        float Cv = xdbl[xbase + (size_t)i * 48 + DTR + DSN + n];
        float dA = __expf(dv * Adn);
        h = dA * h + dv * Bv * uv;
        float prod = h * Cv;
        #pragma unroll
        for (int off = 8; off; off >>= 1) prod += __shfl_xor(prod, off, 16);
        if (n == 0) {
            int spath = scan_path[lbase + i];   // spatial slot for this scan position
            yt[(size_t)((b << 12) + spath) * DI + d] = prod + uv * Dd;
        }
    }
}

// layernorm over DI + *z, one block per (b,l); fully coalesced reads; emits f16
__global__ void ln_mul_kernel(const float* __restrict__ yt, const float* __restrict__ z,
                              const float* __restrict__ lnw, const float* __restrict__ lnb,
                              _Float16* __restrict__ t)
{
    int m = blockIdx.x;
    int d = threadIdx.x;
    float yv = yt[(size_t)m * DI + d];
    float s1 = yv, s2 = yv * yv;
    #pragma unroll
    for (int off = 16; off; off >>= 1) {
        s1 += __shfl_xor(s1, off, 32);
        s2 += __shfl_xor(s2, off, 32);
    }
    __shared__ float p1[12], p2[12];
    __shared__ float smu, srs;
    int wid = d >> 5, lid = d & 31;
    if (lid == 0) { p1[wid] = s1; p2[wid] = s2; }
    __syncthreads();
    if (d == 0) {
        float t1 = 0.0f, t2 = 0.0f;
        #pragma unroll
        for (int i = 0; i < 12; ++i) { t1 += p1[i]; t2 += p2[i]; }
        float mu  = t1 * (1.0f / DI);
        float var = t2 * (1.0f / DI) - mu * mu;
        smu = mu;
        srs = rsqrtf(var + 1e-5f);
    }
    __syncthreads();
    float tv = (yv - smu) * srs * lnw[d] + lnb[d];
    t[(size_t)m * DI + d] = (_Float16)(tv * z[(size_t)m * DI + d]);
}

extern "C" void kernel_launch(void* const* d_in, const int* in_sizes, int n_in,
                              void* d_out, int out_size, void* d_ws, size_t ws_size,
                              hipStream_t stream) {
    (void)in_sizes; (void)n_in; (void)out_size; (void)ws_size;
    const float* x         = (const float*)d_in[0];
    const float* cond      = (const float*)d_in[1];
    const float* W_in      = (const float*)d_in[2];
    const float* W_con     = (const float*)d_in[3];
    const float* conv_w    = (const float*)d_in[4];
    const float* conv_b    = (const float*)d_in[5];
    const float* ccw       = (const float*)d_in[6];
    const float* ccb       = (const float*)d_in[7];
    const float* x_proj_w  = (const float*)d_in[8];
    const float* dt_proj_w = (const float*)d_in[9];
    const float* dt_proj_b = (const float*)d_in[10];
    const float* A_logs    = (const float*)d_in[11];
    const float* Ds        = (const float*)d_in[12];
    const float* ln_w      = (const float*)d_in[13];
    const float* ln_b      = (const float*)d_in[14];
    const float* W_out     = (const float*)d_in[15];
    const int*   scan_path = (const int*)d_in[16];
    float* out = (float*)d_out;

    float* ws = (float*)d_ws;
    const size_t N  = (size_t)BB * DI * LL;          // 3,145,728 floats
    const size_t NX = (size_t)BB * LL * 48;          // 393,216 floats
    const size_t NS = (size_t)BB * DI * NCH * DSN;   // 393,216 floats
    // f32 region
    float* xa_t  = ws;                 // (B,DI,H,W)
    float* con_t = ws + N;             // (B,DI,H,W)
    float* zbuf  = ws + 2 * N;         // (B*L, DI)
    float* ubuf  = ws + 3 * N;         // (b,d,l')
    float* xdbl  = ws + 4 * N;         // (B*L, 48)
    float* Pbuf  = ws + 4 * N + NX;
    float* Sbuf  = Pbuf + NS;
    float* Hbuf  = Sbuf + NS;
    float* delta = xa_t;               // reuse (xa_t dead after conv)
    float* ytbuf = con_t;              // reuse (con_t dead after conv); (B*L, DI) spatial
    // f16 region
    _Float16* h16 = (_Float16*)(Hbuf + NS);
    const size_t NXC = (size_t)BB * LL * DM;         // 1,572,864
    _Float16* xh   = h16;                            // x packed
    _Float16* ch   = xh + NXC;                       // cond packed
    _Float16* wih  = ch + NXC;                       // W_in  (768x192)
    _Float16* wch  = wih + (size_t)(2 * DI) * DM;    // W_con (384x192)
    _Float16* xpjh = wch + (size_t)DI * DM;          // x_proj padded (48x384)
    _Float16* woh  = xpjh + (size_t)48 * DI;         // W_out (192x384)
    _Float16* sh   = woh + (size_t)DM * DI;          // s = xs+cs (B*L, DI)
    _Float16* th   = sh + N;                         // t (B*L, DI)

    const int Mb = (BB * LL) / 64;                   // 128 M-blocks (64 rows each)

    // 0) pack f16 operands
    pack_f16_v4<<<(int)(NXC / 4 / 256), 256, 0, stream>>>(x, xh, (int)(NXC / 4));
    pack_f16_v4<<<(int)(NXC / 4 / 256), 256, 0, stream>>>(cond, ch, (int)(NXC / 4));
    pack_f16_v4<<<(2 * DI * DM / 4) / 256, 256, 0, stream>>>(W_in, wih, 2 * DI * DM / 4);
    pack_f16_v4<<<(DI * DM / 4) / 256, 256, 0, stream>>>(W_con, wch, DI * DM / 4);
    pack_f16_v4<<<(DM * DI / 4) / 256, 256, 0, stream>>>(W_out, woh, DM * DI / 4);
    pack_pad_xproj<<<(48 * DI) / 256, 256, 0, stream>>>(x_proj_w, xpjh);

    // 1) xz = x @ W_in^T  (N=768): xa -> NCHW, z -> silu, (B*L,DI)
    wmma_gemm_kernel<0, DM, DM, DM><<<(Mb * 48) / 4, 128, 0, stream>>>(
        xh, wih, xa_t, zbuf, Mb, 48);
    // 2) c = cond @ W_con^T (N=384) -> NCHW
    wmma_gemm_kernel<1, DM, DM, DM><<<(Mb * 24) / 4, 128, 0, stream>>>(
        ch, wch, con_t, nullptr, Mb, 24);
    // 3) fused dwconv+silu+gather
    conv_scan_kernel<<<(BB * DI * LL) / 256, 256, 0, stream>>>(
        xa_t, con_t, conv_w, conv_b, ccw, ccb, scan_path, ubuf, sh);
    // 4) x_dbl = s @ x_proj_w^T (48-padded)
    wmma_gemm_kernel<2, DI, DI, DI><<<(Mb * 3) / 4, 128, 0, stream>>>(
        sh, xpjh, xdbl, nullptr, Mb, 3);
    // 5) delta = softplus(dt proj)
    dt_kernel<<<(BB * DI * LL) / 256, 256, 0, stream>>>(xdbl, dt_proj_w, dt_proj_b, delta);
    // 6) chunked selective scan (3 passes); pass 3 scatters to spatial (b,l,d)
    scan_chunk_reduce<<<(BB * DI * NCH) / 16, 256, 0, stream>>>(
        ubuf, delta, xdbl, A_logs, Pbuf, Sbuf);
    scan_combine<<<(BB * DI * DSN) / 256, 256, 0, stream>>>(Pbuf, Sbuf, Hbuf);
    scan_apply<<<(BB * DI * NCH) / 16, 256, 0, stream>>>(
        ubuf, delta, xdbl, A_logs, Ds, Hbuf, scan_path, ytbuf);
    // 7) layernorm + *z -> f16 (coalesced)
    ln_mul_kernel<<<BB * LL, DI, 0, stream>>>(ytbuf, zbuf, ln_w, ln_b, th);
    // 8) out = t @ W_out^T
    wmma_gemm_kernel<3, DI, DI, DI><<<(Mb * 12) / 4, 128, 0, stream>>>(
        th, woh, out, nullptr, Mb, 12);
}
